// GCN_5978594476289
// MI455X (gfx1250) — matrix operand order repaired and checked by reading, not compiled, not run on hardware
//
#include <hip/hip_runtime.h>
#include <hip/hip_bf16.h>

typedef __bf16 v16bf __attribute__((ext_vector_type(16)));
typedef float  v8f   __attribute__((ext_vector_type(8)));

#define N_NODES  40000
#define N_EDGES  640000
#define N_GRAPHS 64
#define IN_DIM   512
#define EMB      128

// ---- workspace layout (offsets in floats) ----
// zeroed region first (one memset covers it)
#define OFF_DEGOUT 0                       // 40000
#define OFF_DEGIN  40000                   // 40000
#define OFF_COUNTS 80000                   // 64
#define OFF_POOL1  80064                   // 8192
#define OFF_POOL2  88256                   // 8192
#define OFF_AGG    96448                   // 40000*128 = 5120000
#define ZERO_FLOATS (OFF_AGG + 5120000)    // 5216448
// non-zeroed region
#define OFF_H1     ZERO_FLOATS             // 5120000
#define OFF_Y      (OFF_H1 + 5120000)      // 5120000
#define OFF_W1F    (OFF_Y + 5120000)       // 65536 bf16 = 32768 floats
#define OFF_W2F    (OFF_W1F + 32768)       // 16384 bf16 = 8192 floats

// ------------------------------------------------------------------
// degree / graph-count accumulation
__global__ __launch_bounds__(256) void k_degrees(const int* __restrict__ src,
                                                 const int* __restrict__ dst,
                                                 const int* __restrict__ batch,
                                                 float* __restrict__ deg_out,
                                                 float* __restrict__ deg_in,
                                                 float* __restrict__ counts) {
    int t = blockIdx.x * 256 + threadIdx.x;
    if (t < N_EDGES) {
        atomicAdd(&deg_out[src[t]], 1.0f);
        atomicAdd(&deg_in[dst[t]], 1.0f);
    }
    if (t < N_NODES) {
        atomicAdd(&counts[batch[t]], 1.0f);
    }
}

// deg -> D^{-1/2} in place
__global__ __launch_bounds__(256) void k_norms(float* __restrict__ deg_out,
                                               float* __restrict__ deg_in) {
    int t = blockIdx.x * 256 + threadIdx.x;
    if (t >= N_NODES) return;
    float a = deg_out[t];
    deg_out[t] = (a > 0.0f) ? rsqrtf(fmaxf(a, 1.0f)) : 0.0f;
    float b = deg_in[t];
    deg_in[t] = (b > 0.0f) ? rsqrtf(fmaxf(b, 1.0f)) : 0.0f;
}

// ------------------------------------------------------------------
// Pack W (K x 128 row-major f32) into per-wave WMMA B fragments (bf16).
// Fragment layout for V_WMMA_*_16X16X32_BF16 B operand (32x16, per ISA 7.12.2):
//   lane<16 : column n0+lane,     K = k0+0..7 and k0+16..23
//   lane>=16: column n0+lane-16,  K = k0+8..15 and k0+24..31
// Stored flat: idx = ((ntile*KSTEPS + kstep)*32 + lane)*16 + h
__global__ __launch_bounds__(256) void k_pack_w(const float* __restrict__ W1,
                                                const float* __restrict__ W2,
                                                __bf16* __restrict__ W1f,
                                                __bf16* __restrict__ W2f) {
    int t = blockIdx.x * 256 + threadIdx.x;
    const int F1 = 8 * (IN_DIM / 32) * 32 * 16;  // 65536
    const int F2 = 8 * (EMB   / 32) * 32 * 16;   // 16384
    const float* W;
    __bf16* Wf;
    int ksteps, id;
    if (t < F1)              { W = W1; Wf = W1f; ksteps = IN_DIM / 32; id = t; }
    else if (t < F1 + F2)    { W = W2; Wf = W2f; ksteps = EMB   / 32; id = t - F1; }
    else return;
    int h     = id & 15;
    int lane  = (id >> 4) & 31;
    int rest  = id >> 9;
    int kstep = rest % ksteps;
    int ntile = rest / ksteps;
    int col   = ntile * 16 + (lane & 15);
    int krow  = kstep * 32 + (lane >> 4) * 8 + (h & 7) + (h >> 3) * 16;
    Wf[id] = (__bf16)W[(size_t)krow * EMB + col];
}

// ------------------------------------------------------------------
// Y[M,128] = (X[M,K] * scale[M]) @ W   via v_wmma_f32_16x16x32_bf16.
// Block = 256 threads = 8 waves; wave w owns the 16x16 tile at columns w*16.
// A fragment (16x32, ISA 7.12.2): lane<16 row m0+lane holds K = kb+0..7 and
// kb+16..23; lane>=16 same rows with K = kb+8..15 and kb+24..31.
template <int K>
__global__ __launch_bounds__(256) void k_gemm(const float* __restrict__ X,
                                              const float* __restrict__ scale,
                                              const __bf16* __restrict__ Wf,
                                              float* __restrict__ Y) {
    constexpr int KSTEPS = K / 32;
    const int wave = threadIdx.x >> 5;   // N tile index 0..7
    const int lane = threadIdx.x & 31;
    const int half = lane >> 4;          // 0 or 1
    const int m0   = blockIdx.x * 16;
    const int mrow = m0 + (lane & 15);

    const float* arow = X + (size_t)mrow * K;
    const float  s    = scale[mrow];

    const __bf16* wb = Wf + (size_t)wave * KSTEPS * 512 + (size_t)lane * 16;

    v8f c = {};
    for (int ks = 0; ks < KSTEPS; ++ks) {
        const int kb = ks * 32 + half * 8;
        const float4* p = (const float4*)(arow + kb);
        const float4* q = (const float4*)(arow + kb + 16);
        float4 a0 = p[0], a1 = p[1];
        float4 b0 = q[0], b1 = q[1];
        float f[16];
        f[0]  = a0.x; f[1]  = a0.y; f[2]  = a0.z; f[3]  = a0.w;
        f[4]  = a1.x; f[5]  = a1.y; f[6]  = a1.z; f[7]  = a1.w;
        f[8]  = b0.x; f[9]  = b0.y; f[10] = b0.z; f[11] = b0.w;
        f[12] = b1.x; f[13] = b1.y; f[14] = b1.z; f[15] = b1.w;
        v16bf a;
#pragma unroll
        for (int j = 0; j < 16; ++j) a[j] = (__bf16)(f[j] * s);
        v16bf b = *(const v16bf*)(wb + (size_t)ks * 512);
        c = __builtin_amdgcn_wmma_f32_16x16x32_bf16(
            /*neg_a=*/false, a, /*neg_b=*/false, b,
            /*c_mod=*/(short)0, c, /*reuse_a=*/false, /*reuse_b=*/false);
    }
    // C/D layout: VGPR r -> (M = r + half*8, N = lane&15) within tile
    const int rrow = m0 + half * 8;
    const int col  = wave * 16 + (lane & 15);
#pragma unroll
    for (int r = 0; r < 8; ++r)
        Y[(size_t)(rrow + r) * EMB + col] = c[r];
}

// ------------------------------------------------------------------
// AGG[dst] += Y[src]  — one wave per edge, lane covers 4 dims (coalesced)
__global__ __launch_bounds__(256) void k_scatter(const float* __restrict__ Y,
                                                 const int* __restrict__ src,
                                                 const int* __restrict__ dst,
                                                 float* __restrict__ AGG) {
    int e = blockIdx.x * 8 + (threadIdx.x >> 5);
    if (e >= N_EDGES) return;
    int lane = threadIdx.x & 31;
    int s = src[e], d = dst[e];
    float4 v = *(const float4*)(Y + (size_t)s * EMB + lane * 4);
    float* o = AGG + (size_t)d * EMB + lane * 4;
    atomicAdd(o + 0, v.x);
    atomicAdd(o + 1, v.y);
    atomicAdd(o + 2, v.z);
    atomicAdd(o + 3, v.w);
}

// h = relu(AGG * in_isqrt + bias); optional store; pool accumulation
__global__ __launch_bounds__(256) void k_post(const float* __restrict__ AGG,
                                              const float* __restrict__ in_isqrt,
                                              const float* __restrict__ bias,
                                              const int* __restrict__ batch,
                                              float* __restrict__ H,   // may be null
                                              float* __restrict__ pool) {
    int t = blockIdx.x * 256 + threadIdx.x;
    if (t >= N_NODES * EMB) return;
    int n = t >> 7, d = t & 127;
    float h = fmaxf(AGG[t] * in_isqrt[n] + bias[d], 0.0f);
    if (H) H[t] = h;
    atomicAdd(&pool[(size_t)batch[n] * EMB + d], h);
}

// out = pool1/c + 2*pool2/c
__global__ __launch_bounds__(256) void k_final(const float* __restrict__ pool1,
                                               const float* __restrict__ pool2,
                                               const float* __restrict__ counts,
                                               float* __restrict__ out) {
    int t = blockIdx.x * 256 + threadIdx.x;
    if (t >= N_GRAPHS * EMB) return;
    int g = t >> 7;
    float c = fmaxf(counts[g], 1.0f);
    out[t] = (pool1[t] + 2.0f * pool2[t]) / c;
}

// ------------------------------------------------------------------
extern "C" void kernel_launch(void* const* d_in, const int* in_sizes, int n_in,
                              void* d_out, int out_size, void* d_ws, size_t ws_size,
                              hipStream_t stream) {
    const float* x     = (const float*)d_in[0];
    const int*   src   = (const int*)d_in[1];
    const int*   dst   = (const int*)d_in[2];
    const int*   batch = (const int*)d_in[3];
    const float* W1    = (const float*)d_in[4];
    const float* b1    = (const float*)d_in[5];
    const float* W2    = (const float*)d_in[6];
    const float* b2    = (const float*)d_in[7];
    float* out = (float*)d_out;

    float* ws       = (float*)d_ws;
    float* deg_out  = ws + OFF_DEGOUT;
    float* deg_in   = ws + OFF_DEGIN;
    float* counts   = ws + OFF_COUNTS;
    float* pool1    = ws + OFF_POOL1;
    float* pool2    = ws + OFF_POOL2;
    float* AGG      = ws + OFF_AGG;
    float* H1       = ws + OFF_H1;
    float* Y        = ws + OFF_Y;
    __bf16* W1f     = (__bf16*)(ws + OFF_W1F);
    __bf16* W2f     = (__bf16*)(ws + OFF_W2F);

    // zero accumulation buffers (degrees, counts, pools, AGG)
    hipMemsetAsync(ws, 0, (size_t)ZERO_FLOATS * sizeof(float), stream);

    // degrees + graph counts, then D^{-1/2}
    k_degrees<<<(N_EDGES + 255) / 256, 256, 0, stream>>>(src, dst, batch,
                                                         deg_out, deg_in, counts);
    k_norms<<<(N_NODES + 255) / 256, 256, 0, stream>>>(deg_out, deg_in);

    // pack weights into WMMA bf16 fragments
    k_pack_w<<<(65536 + 16384 + 255) / 256, 256, 0, stream>>>(W1, W2, W1f, W2f);

    // ---- layer 1 ----
    k_gemm<IN_DIM><<<N_NODES / 16, 256, 0, stream>>>(x, deg_out, W1f, Y);
    k_scatter<<<N_EDGES / 8, 256, 0, stream>>>(Y, src, dst, AGG);
    k_post<<<(N_NODES * EMB) / 256, 256, 0, stream>>>(AGG, deg_in, b1, batch, H1, pool1);

    // re-zero AGG for layer 2
    hipMemsetAsync(AGG, 0, (size_t)N_NODES * EMB * sizeof(float), stream);

    // ---- layer 2 ----  (A2 = relu(...)*out_isqrt handled via scale arg)
    k_gemm<EMB><<<N_NODES / 16, 256, 0, stream>>>(H1, deg_out, W2f, Y);
    k_scatter<<<N_EDGES / 8, 256, 0, stream>>>(Y, src, dst, AGG);
    k_post<<<(N_NODES * EMB) / 256, 256, 0, stream>>>(AGG, deg_in, b2, batch, nullptr, pool2);

    // out = skip1 + skip2 + pooled = pool1/c + 2*pool2/c
    k_final<<<(N_GRAPHS * EMB + 255) / 256, 256, 0, stream>>>(pool1, pool2, counts, out);
}
